// RnnGenerateNet_9371618640611
// MI455X (gfx1250) — matrix operand order repaired and checked by reading, not compiled, not run on hardware
//
#include <hip/hip_runtime.h>
#include <hip/hip_bf16.h>

// ---------------------------------------------------------------------------
// Fused 9-layer GRU stack for MI455X (gfx1250, wave32, WMMA f32_16x16x32_f16)
//
// gru0 = 3 layers (in 9/64/64, H=64), gru1 = 6 layers (in 64/9.., H=9).
// B=32768, T=16. One block = 64 batch rows, whole net fused in one kernel;
// HBM traffic is only x-in and sigmoid-out. Weights packed once per launch
// into d_ws in WMMA B-fragment order so LDS B loads are contiguous.
//
// Big layers (H=64): 8 waves split 4 M-tiles x 12 N-tiles, 1-deep software
// pipeline across N-tiles (consume j-1 while issuing j) to cover WMMA->VALU
// hazard slots; block-wide barriers separate GEMM and gate phases.
// Small layers (H=9): one wave owns a 16-row M-tile and ALL 3 gate tiles ->
// gate math is lane-local on the C fragments, h lives in registers, and the
// whole 16-step recurrence runs with ZERO block barriers. One per layer.
// Activations use v_rcp_f32-based sigmoid/tanh (no IEEE divide expansion).
// ---------------------------------------------------------------------------

typedef __attribute__((ext_vector_type(16))) _Float16 v16h;
typedef __attribute__((ext_vector_type(8)))  _Float16 v8h;
typedef __attribute__((ext_vector_type(4)))  _Float16 v4h;
typedef __attribute__((ext_vector_type(8)))  float    v8f;
typedef __attribute__((ext_vector_type(4)))  float    v4f;

#define DEV static __device__ __forceinline__

// Per-layer metadata (9 layers total)
__device__ static const int L_IN [9] = {9, 64, 64, 64, 9, 9, 9, 9, 9};
__device__ static const int L_H  [9] = {64, 64, 64, 9, 9, 9, 9, 9, 9};
__device__ static const int L_HP [9] = {64, 64, 64, 16, 16, 16, 16, 16, 16};
__device__ static const int L_KCX[9] = {1, 2, 2, 2, 1, 1, 1, 1, 1};   // ceil(in/32)
__device__ static const int L_KCH[9] = {2, 2, 2, 1, 1, 1, 1, 1, 1};   // ceil(Hpad/32)
// packed-weight sizes/offsets (halfs): size_l = ntile*(kcx+kch)*512
__device__ static const int L_WSZ [9] = {18432, 24576, 24576, 4608, 3072, 3072, 3072, 3072, 3072};
__device__ static const int L_WOFF[9] = {0, 18432, 43008, 67584, 72192, 75264, 78336, 81408, 84480};
__device__ static const int L_BOFF[9] = {0, 384, 768, 1152, 1248, 1344, 1440, 1536, 1632};

static const int W_TOTAL_HALFS = 87552;  // packed weights
static const int B_TOTAL_FLTS  = 1728;   // packed biases (bx then bh per layer)

struct Params { const float* p[36]; };   // 9 layers x {Wih, Whh, bih, bhh}

// Fast activations: single v_exp_f32 + v_rcp_f32, no IEEE divide expansion.
DEV float fast_sigmoid(float x) {
  return __builtin_amdgcn_rcpf(1.f + __expf(-x));
}
DEV float fast_tanh(float x) {
  return 2.f * __builtin_amdgcn_rcpf(1.f + __expf(-2.f * x)) - 1.f;
}

// ---------------------------------------------------------------------------
// Pack kernel: fp32 weights -> f16 in B-fragment lane order, gate-padded.
// B fragment (32x16): lane = (k_local/16)*16 + n_local, elem e = k_local%16.
// packed[(nt*kc + ks)*512 + lane*16 + e] = W[n][k] with n,k zero-padded.
// ---------------------------------------------------------------------------
__global__ void pack_weights(Params P, _Float16* __restrict__ wpack,
                             float* __restrict__ bpack) {
  const int l   = blockIdx.x;
  const int tid = threadIdx.x;
  const float* Wih = P.p[l * 4 + 0];
  const float* Whh = P.p[l * 4 + 1];
  const float* bih = P.p[l * 4 + 2];
  const float* bhh = P.p[l * 4 + 3];
  const int in  = L_IN[l], h = L_H[l], hp = L_HP[l];
  const int np  = 3 * hp;
  const int kcx = L_KCX[l], kch = L_KCH[l];
  const int ntile = np >> 4;

  _Float16* wx = wpack + L_WOFF[l];
  _Float16* wh = wx + ntile * kcx * 512;

  int total = ntile * kcx * 512;
  for (int i = tid; i < total; i += 256) {
    const int e = i & 15, ln = (i >> 4) & 31, tile = i >> 9;
    const int ks = tile % kcx, nt = tile / kcx;
    const int k = ks * 32 + (ln >> 4) * 16 + e;
    const int n = nt * 16 + (ln & 15);
    const int gate = n / hp, idx = n % hp;
    float v = 0.f;
    if (idx < h && k < in) v = Wih[(gate * h + idx) * in + k];
    wx[i] = (_Float16)v;
  }
  total = ntile * kch * 512;
  for (int i = tid; i < total; i += 256) {
    const int e = i & 15, ln = (i >> 4) & 31, tile = i >> 9;
    const int ks = tile % kch, nt = tile / kch;
    const int k = ks * 32 + (ln >> 4) * 16 + e;
    const int n = nt * 16 + (ln & 15);
    const int gate = n / hp, idx = n % hp;
    float v = 0.f;
    if (idx < h && k < h) v = Whh[(gate * h + idx) * h + k];
    wh[i] = (_Float16)v;
  }
  for (int i = tid; i < np; i += 256) {
    const int gate = i / hp, idx = i % hp;
    float vx = 0.f, vh = 0.f;
    if (idx < h) { vx = bih[gate * h + idx]; vh = bhh[gate * h + idx]; }
    bpack[L_BOFF[l] + i]      = vx;
    bpack[L_BOFF[l] + np + i] = vh;
  }
}

// ---------------------------------------------------------------------------
// Fragment loaders (wave32). 16-bit A 16x32 k order used here (same as B packing):
//   K = half*16 + e  -> two contiguous 16B LDS loads per lane.
// ---------------------------------------------------------------------------
DEV v16h load_a_frag(const _Float16* base, int rstride, int lane) {
  const int half = lane >> 4, m = lane & 15;
  const v8h lo = *(const v8h*)(base + m * rstride + half * 16);
  const v8h hi = *(const v8h*)(base + m * rstride + half * 16 + 8);
  v16h a;
#pragma unroll
  for (int e = 0; e < 8; ++e) { a[e] = lo[e]; a[8 + e] = hi[e]; }
  return a;
}

DEV v16h load_b_frag(const _Float16* frag, int lane) {
  const v8h lo = *(const v8h*)(frag + lane * 16);
  const v8h hi = *(const v8h*)(frag + lane * 16 + 8);
  v16h b;
#pragma unroll
  for (int e = 0; e < 8; ++e) { b[e] = lo[e]; b[8 + e] = hi[e]; }
  return b;
}

DEV v8f wmma_f16(const v16h& a, const v16h& b, const v8f& c) {
  return __builtin_amdgcn_wmma_f32_16x16x32_f16(false, a, false, b, (short)0, c,
                                                false, false);
}

// LDS layout (bytes), TILE_B = 64 rows
static const int SM_SEQ  = 0;                 // 64*16*64 f16  = 131072
static const int SM_H    = 131072;            // 64*64 f16     =   8192
static const int SM_GX   = 139264;            // 64*192 f32    =  49152
static const int SM_GH   = 188416;            // 64*192 f32    =  49152
static const int SM_W    = 237568;            // 24576 f16 max =  49152
static const int SM_BIAS = 286720;            // 384 f32       =   1536
static const int SMEM_BYTES = 288256;         // < 320KB/WGP

// ---------------------------------------------------------------------------
// Big layer (H=64, NTILE=12): 8 waves x (fixed M-tile, 6 N-tiles), pipelined.
// ---------------------------------------------------------------------------
template <int KCX, int KCH>
DEV void run_layer_big(_Float16* s_seq, _Float16* s_h, float* s_gx, float* s_gh,
                       const _Float16* s_w, const float* s_b, int tid, int lane,
                       int wv, int half) {
  constexpr int NP = 192;
  constexpr int NJ = 6;
  const _Float16* s_wx = s_w;
  const _Float16* s_wh = s_w + 12 * KCX * 512;
  const int mt  = wv & 3;
  const int nt0 = (wv >> 2) & 1;  // range-limited so everything folds
  const int nlo = lane & 15;

  // Bias values are loop-invariant over t.
  float bxv[NJ], bhv[NJ];
#pragma unroll
  for (int j = 0; j < NJ; ++j) {
    bxv[j] = s_b[(nt0 + 2 * j) * 16 + nlo];
    bhv[j] = s_b[NP + (nt0 + 2 * j) * 16 + nlo];
  }

  auto consume = [&](const v8f& cxv, const v8f& chv, int j) {
    const int n = (nt0 + 2 * j) * 16 + nlo;
#pragma unroll
    for (int r8 = 0; r8 < 8; ++r8) {  // C layout: m = r8 + 8*half
      const int m = mt * 16 + r8 + 8 * half;
      s_gx[m * NP + n] = cxv[r8] + bxv[j];
      s_gh[m * NP + n] = chv[r8] + bhv[j];
    }
  };

  for (int t = 0; t < 16; ++t) {
    v16h ax[KCX], ah[KCH];
    const _Float16* axp = s_seq + mt * 16 * 1024 + t * 64;
#pragma unroll
    for (int ks = 0; ks < KCX; ++ks)
      ax[ks] = load_a_frag(axp + ks * 32, 1024, lane);
#pragma unroll
    for (int ks = 0; ks < KCH; ++ks)
      ah[ks] = load_a_frag(s_h + mt * 16 * 64 + ks * 32, 64, lane);

    // 1-deep pipeline: consume tile j-1 while issuing tile j.
    v8f pcx = {}, pch = {};
#pragma unroll
    for (int j = 0; j <= NJ; ++j) {
      v8f cx = {}, ch = {};
      if (j < NJ) {
        const int nt = nt0 + 2 * j;
#pragma unroll
        for (int ks = 0; ks < KCX; ++ks)
          cx = wmma_f16(ax[ks], load_b_frag(s_wx + (nt * KCX + ks) * 512, lane), cx);
#pragma unroll
        for (int ks = 0; ks < KCH; ++ks)
          ch = wmma_f16(ah[ks], load_b_frag(s_wh + (nt * KCH + ks) * 512, lane), ch);
      }
      if (j > 0) consume(pcx, pch, j - 1);
      pcx = cx;
      pch = ch;
    }
    __syncthreads();
    // Gate phase: each thread owns 4 consecutive columns -> b128/b64 LDS ops.
#pragma unroll
    for (int c = 0; c < 4; ++c) {
      const int base = tid * 4 + c * 1024;  // element in [64 rows][64 cols]
      const int j = base & 63, r = base >> 6;
      const float* gx = s_gx + r * NP;
      const float* gh = s_gh + r * NP;
      const v4f xr = *(const v4f*)(gx + j);
      const v4f hr = *(const v4f*)(gh + j);
      const v4f xz = *(const v4f*)(gx + 64 + j);
      const v4f hz = *(const v4f*)(gh + 64 + j);
      const v4f xn = *(const v4f*)(gx + 128 + j);
      const v4f hn = *(const v4f*)(gh + 128 + j);
      const v4h hp4 = *(const v4h*)(s_h + base);
      v4h out4;
#pragma unroll
      for (int q = 0; q < 4; ++q) {
        const float rg = fast_sigmoid(xr[q] + hr[q]);
        const float zg = fast_sigmoid(xz[q] + hz[q]);
        const float ng = fast_tanh(xn[q] + rg * hn[q]);
        out4[q] = (_Float16)((1.f - zg) * ng + zg * (float)hp4[q]);
      }
      *(v4h*)(s_h + base) = out4;
      *(v4h*)(s_seq + (r << 10) + (t << 6) + j) = out4;
    }
    __syncthreads();
  }
}

// ---------------------------------------------------------------------------
// Small layer (H=9, NTILE=3, KCH=1, HP=16): waves 0..3 each own a 16-row
// M-tile and all 3 gate tiles. Gate math is lane-local on the C fragments,
// h persists in registers; no block barriers inside the 16-step recurrence.
// ZFILL: also zero seq cols 16..31 (first small layer clears stale gru0 out).
// ---------------------------------------------------------------------------
template <int KCX, bool ZFILL>
DEV void run_layer_small(_Float16* s_seq, _Float16* s_h, const _Float16* s_w,
                         const float* s_b, int lane, int wv, int half) {
  if (wv < 4) {  // wave-uniform; EXEC stays all-ones for active waves
    const int mt  = wv;
    const int nlo = lane & 15;
    const _Float16* s_wx = s_w;
    const _Float16* s_wh = s_w + 3 * KCX * 512;

    float bx[3], bh[3];
#pragma unroll
    for (int g = 0; g < 3; ++g) {
      bx[g] = s_b[g * 16 + nlo];
      bh[g] = s_b[48 + g * 16 + nlo];
    }
    v8f hreg = {};  // h at (m = mt*16 + r8 + 8*half, n = nlo)

    for (int t = 0; t < 16; ++t) {
      v16h axf[KCX];
      const _Float16* axp = s_seq + mt * 16 * 1024 + t * 64;
#pragma unroll
      for (int ks = 0; ks < KCX; ++ks)
        axf[ks] = load_a_frag(axp + ks * 32, 1024, lane);
      // h A-fragment: cols 16..31 stay zero (never written below).
      const v16h ahf = load_a_frag(s_h + mt * 16 * 64, 64, lane);

      v8f gxa[3], gha[3];
#pragma unroll
      for (int g = 0; g < 3; ++g) {
        v8f cx = {};
#pragma unroll
        for (int ks = 0; ks < KCX; ++ks)
          cx = wmma_f16(axf[ks], load_b_frag(s_wx + (g * KCX + ks) * 512, lane), cx);
        gxa[g] = cx;
        v8f ch = {};
        gha[g] = wmma_f16(ahf, load_b_frag(s_wh + g * 512, lane), ch);
      }

      _Float16* hrow = s_h + (mt * 16 + 8 * half) * 64 + nlo;
      _Float16* orow = s_seq + (mt * 16 + 8 * half) * 1024 + t * 64 + nlo;
#pragma unroll
      for (int r8 = 0; r8 < 8; ++r8) {
        const float rg = fast_sigmoid(gxa[0][r8] + bx[0] + gha[0][r8] + bh[0]);
        const float zg = fast_sigmoid(gxa[1][r8] + bx[1] + gha[1][r8] + bh[1]);
        const float ng = fast_tanh(gxa[2][r8] + bx[2] + rg * (gha[2][r8] + bh[2]));
        const float hnew = (1.f - zg) * ng + zg * hreg[r8];
        hreg[r8] = hnew;
        hrow[r8 * 64] = (_Float16)hnew;     // same-wave LDS order -> next A load
        orow[r8 * 1024] = (_Float16)hnew;   // layer output
        if (ZFILL) orow[r8 * 1024 + 16] = (_Float16)0.f;  // clear cols 16..31
      }
    }
  }
  __syncthreads();  // publish seq outputs to all waves (next layer remaps rows)
}

// ---------------------------------------------------------------------------
// Fused GRU kernel: one block = 64 batch rows, 256 threads = 8 wave32.
// ---------------------------------------------------------------------------
__global__ __launch_bounds__(256) void gru_fused(
    const float* __restrict__ x, const _Float16* __restrict__ wpack,
    const float* __restrict__ bpack, float* __restrict__ out) {
  extern __shared__ char smem[];
  _Float16* s_seq = (_Float16*)(smem + SM_SEQ);
  _Float16* s_h   = (_Float16*)(smem + SM_H);
  float*    s_gx  = (float*)(smem + SM_GX);
  float*    s_gh  = (float*)(smem + SM_GH);
  _Float16* s_w   = (_Float16*)(smem + SM_W);
  float*    s_b   = (float*)(smem + SM_BIAS);

  const int tid  = threadIdx.x;
  const int lane = tid & 31;
  const int wv   = tid >> 5;
  const int half = lane >> 4;
  const int b0   = blockIdx.x * 64;

  // Prefetch layer-0 weights while we load x.
  {
    const char* w0 = (const char*)wpack;
    if (tid * 256 < L_WSZ[0] * 2) __builtin_prefetch(w0 + tid * 256, 0, 1);
  }

  // Zero-fill seq tile, then coalesced dense x copy ((b0+r)*144+t*9+c is
  // contiguous in i = r*144 + t*9 + c = rt*9 + c).
  {
    const v8h z8 = {};
    for (int i = tid * 8; i < 64 * 16 * 64; i += 256 * 8) *(v8h*)(s_seq + i) = z8;
  }
  __syncthreads();
  for (int i = tid; i < 64 * 16 * 9; i += 256) {
    const int c = i % 9, rt = i / 9;
    s_seq[rt * 64 + c] = (_Float16)x[(size_t)b0 * 144 + i];
  }

  for (int l = 0; l < 9; ++l) {
    const int wsz = L_WSZ[l];
    const _Float16* gw = wpack + L_WOFF[l];
    const float*    gb = bpack + L_BOFF[l];
    const int np = 3 * L_HP[l];

    __syncthreads();  // previous layer fully consumed s_w / s_b / s_h
    for (int i = tid * 8; i < wsz; i += 256 * 8)
      *(v8h*)(s_w + i) = *(const v8h*)(gw + i);
    for (int i = tid; i < 2 * np; i += 256) s_b[i] = gb[i];
    {
      const v8h z8 = {};
      for (int i = tid * 8; i < 64 * 64; i += 256 * 8) *(v8h*)(s_h + i) = z8;
    }
    // Prefetch next layer's packed weights into L2/L0 while this layer runs.
    if (l < 8) {
      const char* nxt = (const char*)(wpack + L_WOFF[l + 1]);
      if (tid * 256 < L_WSZ[l + 1] * 2) __builtin_prefetch(nxt + tid * 256, 0, 1);
    }
    __syncthreads();

    if (l == 0)
      run_layer_big<1, 2>(s_seq, s_h, s_gx, s_gh, s_w, s_b, tid, lane, wv, half);
    else if (l < 3)
      run_layer_big<2, 2>(s_seq, s_h, s_gx, s_gh, s_w, s_b, tid, lane, wv, half);
    else if (l == 3)
      run_layer_small<2, true>(s_seq, s_h, s_w, s_b, lane, wv, half);
    else
      run_layer_small<1, false>(s_seq, s_h, s_w, s_b, lane, wv, half);
  }

  // Final sigmoid + coalesced store [B,16,9] f32 (last layer ended in barrier)
  for (int i = tid; i < 64 * 16 * 9; i += 256) {
    const int c = i % 9, rt = i / 9;
    const float v = (float)s_seq[rt * 64 + c];
    out[(size_t)b0 * 144 + i] = fast_sigmoid(v);
  }
}

extern "C" void kernel_launch(void* const* d_in, const int* in_sizes, int n_in,
                              void* d_out, int out_size, void* d_ws, size_t ws_size,
                              hipStream_t stream) {
  const float* x = (const float*)d_in[0];
  Params P;
  for (int i = 0; i < 36; ++i) P.p[i] = (const float*)d_in[1 + i];

  _Float16* wpack = (_Float16*)d_ws;
  float*    bpack = (float*)((char*)d_ws + (size_t)W_TOTAL_HALFS * 2);
  float*    out   = (float*)d_out;

  // Repack weights every call (deterministic; d_ws is scratch).
  pack_weights<<<9, 256, 0, stream>>>(P, wpack, bpack);

  (void)hipFuncSetAttribute((const void*)gru_fused,
                            hipFuncAttributeMaxDynamicSharedMemorySize,
                            SMEM_BYTES);

  const int nblk = in_sizes[0] / (16 * 9 * 64);  // BATCH/64 = 512
  gru_fused<<<nblk, 256, SMEM_BYTES, stream>>>(x, wpack, bpack, out);
}
